// GGNNRelReason_28252294873784
// MI455X (gfx1250) — compile-verified, matrix operand order checked by reading, not run on hardware
//
#include <hip/hip_runtime.h>
#include <hip/hip_bf16.h>
#include <math.h>

// Problem constants (match reference)
#define N_IMG   16
#define OBJ_PER 64
#define REL_PER 256
#define NOBJ    (N_IMG * OBJ_PER)   // 1024 objects
#define NREL    (N_IMG * REL_PER)   // 4096 relations
#define NUM_OBJ 151
#define NUM_OBJ_P 160               // padded K for probs@emb (mult of 32)
#define NUM_REL 51
#define NUM_REL_P 128               // padded N for W_rel (mult of BN=128)
#define OBJ_DIM 4096
#define HID     512
#define HID3    (3 * HID)
#define T_ITERS 3
#define IOU_THR 0.3f

typedef __bf16 bf16_t;
typedef __attribute__((ext_vector_type(4)))  __bf16 v4bf;
typedef __attribute__((ext_vector_type(8)))  __bf16 v8bf;
typedef __attribute__((ext_vector_type(16))) __bf16 v16bf;
typedef __attribute__((ext_vector_type(4)))  float  v4f;
typedef __attribute__((ext_vector_type(8)))  float  v8f;

// ---------------------------------------------------------------------------
// Tiled bf16 WMMA GEMM:  C[M,N] = A[M,K] * B[K,N] + bias (+C if accum)
//   A  : [M, lda] row-major bf16, M % 128 == 0, lda % 8 == 0, K % 32 == 0
//   Bt : B pre-transposed to [Npad, ldb] row-major bf16, Npad % 128 == 0
// Buffers are padded so ALL staging loads are unguarded 16-byte vectors.
// Block 256 threads (8 wave32), tile 128x128, BK=32, double-buffered LDS
// with register prefetch + global_prefetch_b8 4 K-steps ahead.
// Each wave computes 32x64: 8 x v_wmma_f32_16x16x32_bf16 per K-step from
// 12 ds_load_b128 (fragments in ISA 16-bit lane order: k = 8h+s, then 16+8h+(s-8)).
// ---------------------------------------------------------------------------
#define BM 128
#define BN 128
#define BK 32
#define PFD 4   // prefetch distance in K-steps

__global__ void __launch_bounds__(256)
wmma_gemm_bf16(const bf16_t* __restrict__ A, const bf16_t* __restrict__ Bt,
               float* __restrict__ C, const float* __restrict__ bias,
               int M, int N, int K, int lda, int ldb, int ldc, int accum)
{
    __shared__ bf16_t As[2][BM][BK];   // 16 KB
    __shared__ bf16_t Bs[2][BN][BK];   // 16 KB

    const int tid  = threadIdx.x;
    const int bm   = blockIdx.y * BM;
    const int bn   = blockIdx.x * BN;
    const int wave = tid >> 5;
    const int lane = tid & 31;
    const int hl   = lane >> 4;        // K-half per ISA 16-bit layout
    const int lr   = lane & 15;        // row/col within fragment
    const int wy   = wave >> 1;        // 0..3 : 32-row strip
    const int wx   = wave & 1;         // 0..1 : 64-col strip

    // staging coords: 8 bf16 (16B) per vector op; 2 chunks each for A and B
    const int rm0 = tid >> 2;          // rows 0..63
    const int rm1 = rm0 + 64;          // rows 64..127
    const int ck  = (tid & 3) * 8;

    v8f acc[2][4] = {};
    const int nsteps = K / BK;

    // prologue: stage K-step 0 into buffer 0
    {
        uint4 a0 = *(const uint4*)(A  + (size_t)(bm + rm0) * lda + ck);
        uint4 a1 = *(const uint4*)(A  + (size_t)(bm + rm1) * lda + ck);
        uint4 b0 = *(const uint4*)(Bt + (size_t)(bn + rm0) * ldb + ck);
        uint4 b1 = *(const uint4*)(Bt + (size_t)(bn + rm1) * ldb + ck);
        *(uint4*)&As[0][rm0][ck] = a0;
        *(uint4*)&As[0][rm1][ck] = a1;
        *(uint4*)&Bs[0][rm0][ck] = b0;
        *(uint4*)&Bs[0][rm1][ck] = b1;
    }
    __syncthreads();

    for (int s = 0; s < nsteps; ++s) {
        const int  cur = s & 1;
        const bool pre = (s + 1) < nsteps;

        // deep prefetch: warm L2/WGP$ PFD K-steps ahead (global_prefetch_b8)
        if (s + PFD < nsteps) {
            const int kp = (s + PFD) * BK;
            __builtin_prefetch(A  + (size_t)(bm + rm0) * lda + kp + ck, 0, 1);
            __builtin_prefetch(A  + (size_t)(bm + rm1) * lda + kp + ck, 0, 1);
            __builtin_prefetch(Bt + (size_t)(bn + rm0) * ldb + kp + ck, 0, 1);
            __builtin_prefetch(Bt + (size_t)(bn + rm1) * ldb + kp + ck, 0, 1);
        }

        // register prefetch of next K-tile
        uint4 pa0, pa1, pb0, pb1;
        if (pre) {
            const int k0 = (s + 1) * BK;
            pa0 = *(const uint4*)(A  + (size_t)(bm + rm0) * lda + k0 + ck);
            pa1 = *(const uint4*)(A  + (size_t)(bm + rm1) * lda + k0 + ck);
            pb0 = *(const uint4*)(Bt + (size_t)(bn + rm0) * ldb + k0 + ck);
            pb1 = *(const uint4*)(Bt + (size_t)(bn + rm1) * ldb + k0 + ck);
        }

        // fragments: two 16B LDS vector loads each
        v16bf afrag[2], bfrag[4];
        #pragma unroll
        for (int i = 0; i < 2; ++i) {
            const bf16_t* ar = &As[cur][wy * 32 + i * 16 + lr][0];
            v8bf lo = *(const v8bf*)(ar + hl * 8);
            v8bf hi = *(const v8bf*)(ar + 16 + hl * 8);
            afrag[i] = __builtin_shufflevector(lo, hi,
                0, 1, 2, 3, 4, 5, 6, 7, 8, 9, 10, 11, 12, 13, 14, 15);
        }
        #pragma unroll
        for (int j = 0; j < 4; ++j) {
            const bf16_t* br = &Bs[cur][wx * 64 + j * 16 + lr][0];
            v8bf lo = *(const v8bf*)(br + hl * 8);
            v8bf hi = *(const v8bf*)(br + 16 + hl * 8);
            bfrag[j] = __builtin_shufflevector(lo, hi,
                0, 1, 2, 3, 4, 5, 6, 7, 8, 9, 10, 11, 12, 13, 14, 15);
        }
        #pragma unroll
        for (int i = 0; i < 2; ++i)
            #pragma unroll
            for (int j = 0; j < 4; ++j)
                acc[i][j] = __builtin_amdgcn_wmma_f32_16x16x32_bf16(
                    false, afrag[i], false, bfrag[j],
                    (short)0, acc[i][j], false, false);

        if (pre) {
            const int nxt = cur ^ 1;
            *(uint4*)&As[nxt][rm0][ck] = pa0;
            *(uint4*)&As[nxt][rm1][ck] = pa1;
            *(uint4*)&Bs[nxt][rm0][ck] = pb0;
            *(uint4*)&Bs[nxt][rm1][ck] = pb1;
        }
        __syncthreads();   // stores(nxt) visible; reads(cur) done before reuse
    }

    // epilogue: C/D layout -> VGPR v : row = v + half*8, col = lane%16
    #pragma unroll
    for (int i = 0; i < 2; ++i) {
        #pragma unroll
        for (int vi = 0; vi < 8; ++vi) {
            const int row = bm + wy * 32 + i * 16 + hl * 8 + vi;
            if (row >= M) continue;
            #pragma unroll
            for (int j = 0; j < 4; ++j) {
                const int col = bn + wx * 64 + j * 16 + lr;
                if (col < N) {
                    float o = acc[i][j][vi];
                    if (bias)  o += bias[col];
                    if (accum) o += C[(size_t)row * ldc + col];
                    C[(size_t)row * ldc + col] = o;
                }
            }
        }
    }
}

// ---------------------------------------------------------------------------
// Conversion / support kernels
// ---------------------------------------------------------------------------

// 4-wide f32 -> bf16 (n must be divisible by 4; all our sizes are)
__global__ void f32_to_bf16_vec4(const float* __restrict__ in,
                                 bf16_t* __restrict__ out, long n4)
{
    long i = (long)blockIdx.x * blockDim.x + threadIdx.x;
    if (i >= n4) return;
    v4f v = *(const v4f*)(in + i * 4);
    v4bf o;
    o[0] = (bf16_t)v[0]; o[1] = (bf16_t)v[1];
    o[2] = (bf16_t)v[2]; o[3] = (bf16_t)v[3];
    *(v4bf*)(out + i * 4) = o;
}

// out[n*Kp + k] = (n<N && k<K) ? bf16(in[k*N + n]) : 0   (transpose + pad)
__global__ void cvt_transpose_pad(const float* __restrict__ in,
                                  bf16_t* __restrict__ out,
                                  int K, int N, int Kp, int Np)
{
    long idx = (long)blockIdx.x * blockDim.x + threadIdx.x;
    if (idx >= (long)Np * Kp) return;
    int n = (int)(idx / Kp);
    int k = (int)(idx % Kp);
    out[idx] = (n < N && k < K) ? (bf16_t)in[(size_t)k * N + n] : (bf16_t)0.0f;
}

__global__ void zero_f32_kernel(float* __restrict__ p, long n)
{
    long i = (long)blockIdx.x * blockDim.x + threadIdx.x;
    if (i < n) p[i] = 0.0f;
}

__global__ void copy_f32_kernel(const float* __restrict__ in,
                                float* __restrict__ out, long n)
{
    long i = (long)blockIdx.x * blockDim.x + threadIdx.x;
    if (i < n) out[i] = in[i];
}

// Row softmax over NUM_OBJ classes -> f32 probs [NOBJ,NUM_OBJ] and
// bf16 probs padded to row stride NUM_OBJ_P (pad cols zeroed).
__global__ void __launch_bounds__(256)
softmax_kernel(const float* __restrict__ logits,
               float* __restrict__ probs, bf16_t* __restrict__ probs_bf)
{
    __shared__ float red[256];
    const int row = blockIdx.x;
    const int tid = threadIdx.x;
    float v = (tid < NUM_OBJ) ? logits[(size_t)row * NUM_OBJ + tid] : -3.4e38f;
    red[tid] = v;
    __syncthreads();
    for (int s = 128; s > 0; s >>= 1) {
        if (tid < s) red[tid] = fmaxf(red[tid], red[tid + s]);
        __syncthreads();
    }
    float mx = red[0];
    __syncthreads();
    float e = (tid < NUM_OBJ) ? __expf(v - mx) : 0.0f;
    red[tid] = e;
    __syncthreads();
    for (int s = 128; s > 0; s >>= 1) {
        if (tid < s) red[tid] += red[tid + s];
        __syncthreads();
    }
    float inv = 1.0f / red[0];
    if (tid < NUM_OBJ) {
        float p = e * inv;
        probs[(size_t)row * NUM_OBJ + tid]      = p;
        probs_bf[(size_t)row * NUM_OBJ_P + tid] = (bf16_t)p;
    } else if (tid < NUM_OBJ_P) {
        probs_bf[(size_t)row * NUM_OBJ_P + tid] = (bf16_t)0.0f;
    }
}

// msum[r] = bf16(h_o[sub[r]] + h_o[obj[r]])
__global__ void gather_sum_kernel(const float* __restrict__ h_o,
                                  const int* __restrict__ rel_inds,
                                  bf16_t* __restrict__ msum)
{
    long idx = (long)blockIdx.x * blockDim.x + threadIdx.x;
    if (idx >= (long)NREL * HID) return;
    int rrow = (int)(idx / HID);
    int j    = (int)(idx % HID);
    int s = rel_inds[rrow * 3 + 1];
    int o = rel_inds[rrow * 3 + 2];
    msum[idx] = (bf16_t)(h_o[(size_t)s * HID + j] + h_o[(size_t)o * HID + j]);
}

// m_o[sub] += m_back ; m_o[obj] += m_back (atomic f32 scatter-add)
__global__ void scatter_add_kernel(const float* __restrict__ m_back,
                                   const int* __restrict__ rel_inds,
                                   float* __restrict__ m_o)
{
    long idx = (long)blockIdx.x * blockDim.x + threadIdx.x;
    if (idx >= (long)NREL * HID) return;
    int rrow = (int)(idx / HID);
    int j    = (int)(idx % HID);
    int s = rel_inds[rrow * 3 + 1];
    int o = rel_inds[rrow * 3 + 2];
    float v = m_back[idx];
    atomicAdd(&m_o[(size_t)s * HID + j], v);
    atomicAdd(&m_o[(size_t)o * HID + j], v);
}

// GRU update: h = (1-z)*n + z*h, gates split from gx/gh (rows x 3*HID)
__global__ void gru_kernel(float* __restrict__ h,
                           const float* __restrict__ gx,
                           const float* __restrict__ gh, int rows)
{
    long idx = (long)blockIdx.x * blockDim.x + threadIdx.x;
    if (idx >= (long)rows * HID) return;
    int i = (int)(idx / HID);
    int j = (int)(idx % HID);
    size_t b = (size_t)i * HID3 + j;
    float xz = gx[b], xr = gx[b + HID], xn = gx[b + 2 * HID];
    float hz = gh[b], hr = gh[b + HID], hn = gh[b + 2 * HID];
    float z = 1.0f / (1.0f + __expf(-(xz + hz)));
    float r = 1.0f / (1.0f + __expf(-(xr + hr)));
    float n = tanhf(xn + r * hn);
    float hv = h[idx];
    h[idx] = (1.0f - z) * n + z * hv;
}

// feat = concat(h_o[sub], h_r, h_o[obj]) -> bf16 (NREL x 3*HID)
__global__ void feat_kernel(const float* __restrict__ h_o,
                            const float* __restrict__ h_r,
                            const int* __restrict__ rel_inds,
                            bf16_t* __restrict__ feat)
{
    long idx = (long)blockIdx.x * blockDim.x + threadIdx.x;
    if (idx >= (long)NREL * HID3) return;
    int rrow = (int)(idx / HID3);
    int j    = (int)(idx % HID3);
    float v;
    if (j < HID) {
        v = h_o[(size_t)rel_inds[rrow * 3 + 1] * HID + j];
    } else if (j < 2 * HID) {
        v = h_r[(size_t)rrow * HID + (j - HID)];
    } else {
        v = h_o[(size_t)rel_inds[rrow * 3 + 2] * HID + (j - 2 * HID)];
    }
    feat[idx] = (bf16_t)v;
}

// Per-class greedy NMS (one block per class): stable descending rank sort,
// then sequential suppression over sorted order.
__global__ void __launch_bounds__(256)
nms_kernel(const float* __restrict__ probs,   // [NOBJ, NUM_OBJ]
           const float* __restrict__ boxes,   // [NOBJ, NUM_OBJ, 4]
           float* __restrict__ mask)          // [NOBJ, NUM_OBJ]
{
    const int c   = blockIdx.x;
    const int tid = threadIdx.x;
    __shared__ float         sc[NOBJ];
    __shared__ float         bx[NOBJ][4];
    __shared__ int           ord[NOBJ];
    __shared__ unsigned char keep[NOBJ];
    __shared__ int           flag;

    for (int i = tid; i < NOBJ; i += 256)
        sc[i] = probs[(size_t)i * NUM_OBJ + c];
    __syncthreads();

    for (int i = tid; i < NOBJ; i += 256) {
        float si = sc[i];
        int rank = 0;
        for (int j = 0; j < NOBJ; ++j) {
            float sj = sc[j];
            rank += (sj > si) || (sj == si && j < i);
        }
        ord[rank] = i;
    }
    __syncthreads();

    for (int i = tid; i < NOBJ; i += 256) {
        int src = ord[i];
        const float* b = boxes + ((size_t)src * NUM_OBJ + c) * 4;
        bx[i][0] = b[0]; bx[i][1] = b[1]; bx[i][2] = b[2]; bx[i][3] = b[3];
        keep[i] = 0;
    }
    __syncthreads();

    for (int i = 0; i < NOBJ; ++i) {
        if (tid == 0) flag = 0;
        __syncthreads();
        float x1 = bx[i][0], y1 = bx[i][1], x2 = bx[i][2], y2 = bx[i][3];
        float ai = (x2 - x1) * (y2 - y1);
        int found = 0;
        for (int j = tid; j < i; j += 256) {
            if (keep[j]) {
                float iw = fmaxf(fminf(x2, bx[j][2]) - fmaxf(x1, bx[j][0]), 0.0f);
                float ih = fmaxf(fminf(y2, bx[j][3]) - fmaxf(y1, bx[j][1]), 0.0f);
                float inter = iw * ih;
                float aj = (bx[j][2] - bx[j][0]) * (bx[j][3] - bx[j][1]);
                float iou = inter / (ai + aj - inter);
                if (iou > IOU_THR) { found = 1; break; }
            }
        }
        if (found) flag = 1;   // benign race: all writers store 1
        __syncthreads();
        if (tid == 0) keep[i] = (flag == 0);
        __syncthreads();
    }

    for (int i = tid; i < NOBJ; i += 256) {
        int orig = ord[i];
        mask[(size_t)orig * NUM_OBJ + c] = (keep[i] && c != 0) ? 1.0f : 0.0f;
    }
}

// obj_preds[i] = argmax over classes 1..NUM_OBJ-1 of mask*prob
__global__ void argmax_kernel(const float* __restrict__ probs,
                              const float* __restrict__ mask,
                              float* __restrict__ preds)
{
    int i = blockIdx.x * blockDim.x + threadIdx.x;
    if (i >= NOBJ) return;
    float best = -1.0f;
    int bc = 1;
    for (int cc = 1; cc < NUM_OBJ; ++cc) {
        float v = probs[(size_t)i * NUM_OBJ + cc] * mask[(size_t)i * NUM_OBJ + cc];
        if (v > best) { best = v; bc = cc; }
    }
    preds[i] = (float)bc;
}

// ---------------------------------------------------------------------------
// Host orchestration
// ---------------------------------------------------------------------------
static inline dim3 gemm_grid(int M, int N)
{
    return dim3((unsigned)((N + BN - 1) / BN), (unsigned)((M + BM - 1) / BM), 1);
}
static inline unsigned gsz(long n) { return (unsigned)((n + 255) / 256); }

extern "C" void kernel_launch(void* const* d_in, const int* in_sizes, int n_in,
                              void* d_out, int out_size, void* d_ws, size_t ws_size,
                              hipStream_t stream)
{
    (void)in_sizes; (void)n_in; (void)out_size; (void)ws_size;

    const float* obj_fmaps  = (const float*)d_in[1];
    const float* obj_logits = (const float*)d_in[2];
    const int*   rel_inds   = (const int*)  d_in[3];
    const float* vr         = (const float*)d_in[4];
    const float* boxes      = (const float*)d_in[5];
    const float* obj_proj_w = (const float*)d_in[6];
    const float* obj_proj_b = (const float*)d_in[7];
    const float* rel_proj_w = (const float*)d_in[8];
    const float* rel_proj_b = (const float*)d_in[9];
    const float* emb        = (const float*)d_in[10];
    const float* W_o2r      = (const float*)d_in[11];
    const float* W_r2o      = (const float*)d_in[12];
    const float* gru_rel_Wx = (const float*)d_in[13];
    const float* gru_rel_Wh = (const float*)d_in[14];
    const float* gru_rel_b  = (const float*)d_in[15];
    const float* gru_obj_Wx = (const float*)d_in[16];
    const float* gru_obj_Wh = (const float*)d_in[17];
    const float* gru_obj_b  = (const float*)d_in[18];
    const float* W_rel      = (const float*)d_in[19];
    const float* b_rel      = (const float*)d_in[20];

    float* out_logits = (float*)d_out;                           // NOBJ*NUM_OBJ
    float* out_preds  = out_logits + (size_t)NOBJ * NUM_OBJ;     // NOBJ
    float* out_rel    = out_preds + NOBJ;                        // NREL*NUM_REL

    // bump allocator over workspace
    char*  base = (char*)d_ws;
    size_t off  = 0;
    auto alloc = [&](size_t bytes) -> void* {
        void* p = base + off;
        off += (bytes + 255) & ~(size_t)255;
        return p;
    };

    // bf16 activations (A matrices, row-major [M, K])
    bf16_t* fmaps_bf = (bf16_t*)alloc((size_t)NOBJ * OBJ_DIM * 2);
    bf16_t* vr_bf    = (bf16_t*)alloc((size_t)NREL * OBJ_DIM * 2);
    bf16_t* probs_bf = (bf16_t*)alloc((size_t)NOBJ * NUM_OBJ_P * 2);
    bf16_t* h_o_bf   = (bf16_t*)alloc((size_t)NOBJ * HID * 2);
    bf16_t* h_r_bf   = (bf16_t*)alloc((size_t)NREL * HID * 2);
    bf16_t* msum_bf  = (bf16_t*)alloc((size_t)NREL * HID * 2);
    bf16_t* m_r_bf   = (bf16_t*)alloc((size_t)NREL * HID * 2);
    bf16_t* m_o_bf   = (bf16_t*)alloc((size_t)NOBJ * HID * 2);
    bf16_t* feat_bf  = (bf16_t*)alloc((size_t)NREL * HID3 * 2);

    // bf16 weights in [N, K] layout (already-transposed ones plain-converted,
    // [K, N] ones transpose-converted with padding)
    bf16_t* objw_bf  = (bf16_t*)alloc((size_t)HID * OBJ_DIM * 2);      // [512,4096]
    bf16_t* relw_bf  = (bf16_t*)alloc((size_t)HID * OBJ_DIM * 2);      // [512,4096]
    bf16_t* emb_t    = (bf16_t*)alloc((size_t)HID * NUM_OBJ_P * 2);    // [512,160]
    bf16_t* Wo2r_t   = (bf16_t*)alloc((size_t)HID * HID * 2);          // [512,512]
    bf16_t* Wr2o_t   = (bf16_t*)alloc((size_t)HID * HID * 2);          // [512,512]
    bf16_t* gRWx_t   = (bf16_t*)alloc((size_t)HID3 * HID * 2);         // [1536,512]
    bf16_t* gRWh_t   = (bf16_t*)alloc((size_t)HID3 * HID * 2);
    bf16_t* gOWx_t   = (bf16_t*)alloc((size_t)HID3 * HID * 2);
    bf16_t* gOWh_t   = (bf16_t*)alloc((size_t)HID3 * HID * 2);
    bf16_t* Wrel_t   = (bf16_t*)alloc((size_t)NUM_REL_P * HID3 * 2);   // [128,1536]

    // f32 buffers
    float* probs  = (float*)alloc((size_t)NOBJ * NUM_OBJ * 4);
    float* h_o    = (float*)alloc((size_t)NOBJ * HID * 4);
    float* h_r    = (float*)alloc((size_t)NREL * HID * 4);
    float* m_r    = (float*)alloc((size_t)NREL * HID * 4);
    float* gx     = (float*)alloc((size_t)NREL * HID3 * 4);
    float* gh     = (float*)alloc((size_t)NREL * HID3 * 4);
    float* m_back = (float*)alloc((size_t)NREL * HID * 4);
    float* m_o    = (float*)alloc((size_t)NOBJ * HID * 4);
    float* nms_mask = (float*)alloc((size_t)NOBJ * NUM_OBJ * 4);

    auto cvt = [&](const float* src, bf16_t* dst, long n) {
        f32_to_bf16_vec4<<<gsz(n / 4), 256, 0, stream>>>(src, dst, n / 4);
    };
    auto cvtT = [&](const float* src, bf16_t* dst, int K, int N, int Kp, int Np) {
        cvt_transpose_pad<<<gsz((long)Np * Kp), 256, 0, stream>>>(src, dst, K, N, Kp, Np);
    };

    // One-time (per call) bf16 conversions
    cvt(obj_fmaps,  fmaps_bf, (long)NOBJ * OBJ_DIM);
    cvt(vr,         vr_bf,    (long)NREL * OBJ_DIM);
    cvt(obj_proj_w, objw_bf,  (long)HID * OBJ_DIM);
    cvt(rel_proj_w, relw_bf,  (long)HID * OBJ_DIM);
    cvtT(emb,        emb_t,  NUM_OBJ, HID,  NUM_OBJ_P, HID);
    cvtT(W_o2r,      Wo2r_t, HID,     HID,  HID,       HID);
    cvtT(W_r2o,      Wr2o_t, HID,     HID,  HID,       HID);
    cvtT(gru_rel_Wx, gRWx_t, HID,     HID3, HID,       HID3);
    cvtT(gru_rel_Wh, gRWh_t, HID,     HID3, HID,       HID3);
    cvtT(gru_obj_Wx, gOWx_t, HID,     HID3, HID,       HID3);
    cvtT(gru_obj_Wh, gOWh_t, HID,     HID3, HID,       HID3);
    cvtT(W_rel,      Wrel_t, HID3, NUM_REL, HID3, NUM_REL_P);

    // softmax(obj_logits) -> probs (f32) + padded bf16
    softmax_kernel<<<NOBJ, 256, 0, stream>>>(obj_logits, probs, probs_bf);

    // h_o = obj_fmaps @ obj_proj_w.T + b
    wmma_gemm_bf16<<<gemm_grid(NOBJ, HID), 256, 0, stream>>>(
        fmaps_bf, objw_bf, h_o, obj_proj_b,
        NOBJ, HID, OBJ_DIM, OBJ_DIM, OBJ_DIM, HID, 0);
    // h_r = vr @ rel_proj_w.T + b
    wmma_gemm_bf16<<<gemm_grid(NREL, HID), 256, 0, stream>>>(
        vr_bf, relw_bf, h_r, rel_proj_b,
        NREL, HID, OBJ_DIM, OBJ_DIM, OBJ_DIM, HID, 0);
    // h_o += obj_probs @ emb   (K padded 151 -> 160 with zeros)
    wmma_gemm_bf16<<<gemm_grid(NOBJ, HID), 256, 0, stream>>>(
        probs_bf, emb_t, h_o, nullptr,
        NOBJ, HID, NUM_OBJ_P, NUM_OBJ_P, NUM_OBJ_P, HID, 1);

    for (int t = 0; t < T_ITERS; ++t) {
        cvt(h_o, h_o_bf, (long)NOBJ * HID);

        // m_r = (h_o[sub] + h_o[obj]) @ W_o2r
        gather_sum_kernel<<<gsz((long)NREL * HID), 256, 0, stream>>>(
            h_o, rel_inds, msum_bf);
        wmma_gemm_bf16<<<gemm_grid(NREL, HID), 256, 0, stream>>>(
            msum_bf, Wo2r_t, m_r, nullptr, NREL, HID, HID, HID, HID, HID, 0);

        // rel GRU gates
        cvt(m_r, m_r_bf, (long)NREL * HID);
        wmma_gemm_bf16<<<gemm_grid(NREL, HID3), 256, 0, stream>>>(
            m_r_bf, gRWx_t, gx, gru_rel_b, NREL, HID3, HID, HID, HID, HID3, 0);
        cvt(h_r, h_r_bf, (long)NREL * HID);
        wmma_gemm_bf16<<<gemm_grid(NREL, HID3), 256, 0, stream>>>(
            h_r_bf, gRWh_t, gh, nullptr, NREL, HID3, HID, HID, HID, HID3, 0);
        gru_kernel<<<gsz((long)NREL * HID), 256, 0, stream>>>(h_r, gx, gh, NREL);

        // m_back = h_r @ W_r2o ; scatter-add into m_o
        cvt(h_r, h_r_bf, (long)NREL * HID);
        wmma_gemm_bf16<<<gemm_grid(NREL, HID), 256, 0, stream>>>(
            h_r_bf, Wr2o_t, m_back, nullptr, NREL, HID, HID, HID, HID, HID, 0);
        zero_f32_kernel<<<gsz((long)NOBJ * HID), 256, 0, stream>>>(
            m_o, (long)NOBJ * HID);
        scatter_add_kernel<<<gsz((long)NREL * HID), 256, 0, stream>>>(
            m_back, rel_inds, m_o);

        // obj GRU gates (h_o_bf holds h_o entering this iteration)
        cvt(m_o, m_o_bf, (long)NOBJ * HID);
        wmma_gemm_bf16<<<gemm_grid(NOBJ, HID3), 256, 0, stream>>>(
            m_o_bf, gOWx_t, gx, gru_obj_b, NOBJ, HID3, HID, HID, HID, HID3, 0);
        wmma_gemm_bf16<<<gemm_grid(NOBJ, HID3), 256, 0, stream>>>(
            h_o_bf, gOWh_t, gh, nullptr, NOBJ, HID3, HID, HID, HID, HID3, 0);
        gru_kernel<<<gsz((long)NOBJ * HID), 256, 0, stream>>>(h_o, gx, gh, NOBJ);
    }

    // rel_logits = concat(h_o[sub], h_r, h_o[obj]) @ W_rel + b_rel
    feat_kernel<<<gsz((long)NREL * HID3), 256, 0, stream>>>(
        h_o, h_r, rel_inds, feat_bf);
    wmma_gemm_bf16<<<gemm_grid(NREL, NUM_REL_P), 256, 0, stream>>>(
        feat_bf, Wrel_t, out_rel, b_rel,
        NREL, NUM_REL, HID3, HID3, HID3, NUM_REL, 0);

    // obj_logits passthrough
    copy_f32_kernel<<<gsz((long)NOBJ * NUM_OBJ), 256, 0, stream>>>(
        obj_logits, out_logits, (long)NOBJ * NUM_OBJ);

    // per-class NMS + masked argmax -> obj_preds
    nms_kernel<<<NUM_OBJ, 256, 0, stream>>>(probs, boxes, nms_mask);
    argmax_kernel<<<gsz(NOBJ), 256, 0, stream>>>(probs, nms_mask, out_preds);
}
